// MaskHeadNetwork_46961172414629
// MI455X (gfx1250) — matrix operand order repaired and checked by reading, not compiled, not run on hardware
//
#include <hip/hip_runtime.h>

// CondInst-style per-instance dynamic mask head (16->8 relu -> 8->8 relu -> 8->1),
// fused. Memory-bound (~223MB @ 23.3TB/s => ~10us floor): stream pixel embeddings
// once (NT b128), run the GEMM chain on v_wmma_f32_16x16x32_f16 in a transposed
// formulation (out^T = W^T x X^T) so each WMMA's D layout (pixels-in-lanes,
// channels-in-VGPRs) is directly the next layer's B operand layout.
//
// Key operand tricks (vs naive):
//  * Combined layer-0 B: lanes 0-15 = pixel L embs as K=0..15, lanes 16-31 =
//    pixel L+16 embs as K=16..31 (exactly what each lane loaded). Two A variants
//    (W0^T in low-K / high-K half) select tile0/tile1 -> no shuffles, no masks.
//  * Biases ride in the WMMA C operand (rows 0..7 on lanes<16, zero elsewhere),
//    so lanes>=16 of D are exactly zero and layer-1's zero padding is free.
//  * v_cvt_pkrtz_f16_f32 for all f32->f16 packing.

typedef __attribute__((ext_vector_type(16))) _Float16 v16h;
typedef __attribute__((ext_vector_type(2)))  __fp16   fp16x2;
typedef __attribute__((ext_vector_type(8)))  float    v8f;
typedef __attribute__((ext_vector_type(4)))  float    f32x4;

#define NUM_INST   200
#define HW         (128 * 128)
#define PE         16
#define CH         8
#define NUM_PARAMS 217
// param layout (per reference): w0[16*8], b0[8], w1[8*8], b1[8], w2[8], b2[1]
#define OFF_W0 0
#define OFF_B0 128
#define OFF_W1 136
#define OFF_B1 200
#define OFF_W2 208
#define OFF_B2 216

#define WAVES_PER_BLOCK 8
#define ITERS_PER_BLOCK 8
#define PX_PER_BLOCK (WAVES_PER_BLOCK * 32 * ITERS_PER_BLOCK) // 2048

union HV { v16h h; int i[8]; };

__device__ __forceinline__ int pack2(float a, float b) {
    union { fp16x2 h; int i; } u;
    u.h = __builtin_amdgcn_cvt_pkrtz(a, b);   // v_cvt_pkrtz_f16_f32
    return u.i;
}

__global__ __launch_bounds__(256) void mask_head_wmma_kernel(
    const float* __restrict__ params,
    const float* __restrict__ pix,
    float* __restrict__ out)
{
    const int inst  = blockIdx.y;
    const int chunk = blockIdx.x;
    const int lane  = threadIdx.x & 31;
    const int wave  = threadIdx.x >> 5;
    const int r     = lane & 15;        // matrix row within half-wave
    const bool hi   = lane >= 16;

    const float* __restrict__ p = params + inst * NUM_PARAMS;

    // ---- A operands in WMMA 16x32 f16 A-layout ----
    // A layout (16-bit): lane<16 row M=lane: elems 0..7 = K 0..7, elems 8..15 = K 16..23
    //                    lane>=16 row M=lane-16: elems 0..7 = K 8..15, elems 8..15 = K 24..31
    // aw0lo: W0^T in K=0..15 (tile0 selector). aw0hi: W0^T in K=16..31 (tile1 selector).
    // aw1:   W1^T in K=0..7 (layer-1 reduction depth is 8).
    v16h aw0lo, aw0hi, aw1;
#pragma unroll
    for (int j = 0; j < 16; ++j) {
        aw0lo[j] = (_Float16)0.0f; aw0hi[j] = (_Float16)0.0f; aw1[j] = (_Float16)0.0f;
    }
    if (r < CH) {
#pragma unroll
        for (int j = 0; j < 16; ++j) {
            // W0^T[r, c] = w0[c*CH + r]
            const _Float16 w = (_Float16)p[OFF_W0 + j * CH + r];
            if (!hi) aw0lo[j] = w;      // K = j        (lanes<16)
            else     aw0hi[j] = w;      // K = 16 + j   (lanes>=16)
        }
        if (!hi) {
#pragma unroll
            for (int j = 0; j < 8; ++j) {
                aw1[j] = (_Float16)p[OFF_W1 + j * CH + r];   // W1^T[r, c=j]
            }
        }
    }

    // ---- Bias C operands: C[M=i, N] = b[i] on valid rows (lanes<16), else 0.
    // With zero A rows 8..15 this makes lanes>=16 of D exactly zero -> layer-1
    // zero padding needs no masking.
    v8f c0bias, c1bias;
#pragma unroll
    for (int i = 0; i < CH; ++i) {
        c0bias[i] = hi ? 0.f : p[OFF_B0 + i];
        c1bias[i] = hi ? 0.f : p[OFF_B1 + i];
    }

    float w2v[CH];
#pragma unroll
    for (int i = 0; i < CH; ++i) w2v[i] = p[OFF_W2 + i];
    const float b2 = p[OFF_B2];

    const long instPixBase  = (long)inst * HW;
    const int  blockPixBase = chunk * PX_PER_BLOCK;

    for (int it = 0; it < ITERS_PER_BLOCK; ++it) {
        const int  pixInBlock = blockPixBase + it * (WAVES_PER_BLOCK * 32) + wave * 32 + lane;
        const long pixIdx     = instPixBase + pixInBlock;

        // ---- Stream one pixel per lane: 16 f32 = 4 x b128 (non-temporal) ----
        const f32x4* __restrict__ src = (const f32x4*)(pix + pixIdx * PE);
        f32x4 x0 = __builtin_nontemporal_load(src + 0);
        f32x4 x1 = __builtin_nontemporal_load(src + 1);
        f32x4 x2 = __builtin_nontemporal_load(src + 2);
        f32x4 x3 = __builtin_nontemporal_load(src + 3);
        __builtin_prefetch((const char*)src + (size_t)WAVES_PER_BLOCK * 32 * PE * 4, 0, 1);

        // ---- Combined layer-0 B operand: element j = K (lane<16: K=j = emb j of
        // pixel L; lane>=16: K=16+j = emb j of pixel L+16). All real data, no pads.
        HV xh;
        xh.i[0] = pack2(x0.x, x0.y);  xh.i[1] = pack2(x0.z, x0.w);
        xh.i[2] = pack2(x1.x, x1.y);  xh.i[3] = pack2(x1.z, x1.w);
        xh.i[4] = pack2(x2.x, x2.y);  xh.i[5] = pack2(x2.z, x2.w);
        xh.i[6] = pack2(x3.x, x3.y);  xh.i[7] = pack2(x3.z, x3.w);

        // ---- Layer 0: two WMMAs select tile0/tile1 via low/high-K weight copies ----
        v8f d0 = __builtin_amdgcn_wmma_f32_16x16x32_f16(false, aw0lo, false, xh.h, (short)0, c0bias, false, false);
        v8f d1 = __builtin_amdgcn_wmma_f32_16x16x32_f16(false, aw0hi, false, xh.h, (short)0, c0bias, false, false);

        // ---- ReLU + pack as layer-1 B operands (channels become K=0..7).
        // Lanes>=16 hold exact zeros already (zero A rows + zero C) -> pads free.
        HV hb0, hb1;
        hb0.i[0] = pack2(fmaxf(d0[0], 0.f), fmaxf(d0[1], 0.f));
        hb0.i[1] = pack2(fmaxf(d0[2], 0.f), fmaxf(d0[3], 0.f));
        hb0.i[2] = pack2(fmaxf(d0[4], 0.f), fmaxf(d0[5], 0.f));
        hb0.i[3] = pack2(fmaxf(d0[6], 0.f), fmaxf(d0[7], 0.f));
        hb1.i[0] = pack2(fmaxf(d1[0], 0.f), fmaxf(d1[1], 0.f));
        hb1.i[1] = pack2(fmaxf(d1[2], 0.f), fmaxf(d1[3], 0.f));
        hb1.i[2] = pack2(fmaxf(d1[4], 0.f), fmaxf(d1[5], 0.f));
        hb1.i[3] = pack2(fmaxf(d1[6], 0.f), fmaxf(d1[7], 0.f));
#pragma unroll
        for (int k = 4; k < 8; ++k) { hb0.i[k] = 0; hb1.i[k] = 0; }

        // ---- Layer 1 (bias via C operand) ----
        v8f e0 = __builtin_amdgcn_wmma_f32_16x16x32_f16(false, aw1, false, hb0.h, (short)0, c1bias, false, false);
        v8f e1 = __builtin_amdgcn_wmma_f32_16x16x32_f16(false, aw1, false, hb1.h, (short)0, c1bias, false, false);

        // ---- Layer 2: per-lane 8-wide dot (channels live in this lane's VGPRs) ----
        float f0 = b2, f1 = b2;
#pragma unroll
        for (int i = 0; i < CH; ++i) {
            f0 = fmaf(fmaxf(e0[i], 0.f), w2v[i], f0);
            f1 = fmaf(fmaxf(e1[i], 0.f), w2v[i], f1);
        }
        // tile1 results live in lanes 0-15; swap halves for one coalesced 128B store
        const float f1s = __shfl_xor(f1, 16, 32);
        const float val = hi ? f1s : f0;
        __builtin_nontemporal_store(val, out + pixIdx);
    }
}

extern "C" void kernel_launch(void* const* d_in, const int* in_sizes, int n_in,
                              void* d_out, int out_size, void* d_ws, size_t ws_size,
                              hipStream_t stream) {
    const float* params = (const float*)d_in[0];   // [200, 217] f32
    const float* pix    = (const float*)d_in[1];   // [200, 128, 128, 16] f32
    float* out          = (float*)d_out;           // [200, 128, 128] f32
    (void)in_sizes; (void)n_in; (void)out_size; (void)d_ws; (void)ws_size;

    dim3 grid(HW / PX_PER_BLOCK, NUM_INST);        // (8, 200) = 1600 blocks
    dim3 block(WAVES_PER_BLOCK * 32);              // 256 threads = 8 wave32
    hipLaunchKernelGGL(mask_head_wmma_kernel, grid, block, 0, stream, params, pix, out);
}
